// deeds2d_31464930410679
// MI455X (gfx1250) — matrix operand order repaired and checked by reading, not compiled, not run on hardware
//
#include <hip/hip_runtime.h>

#define C_      128
#define H_      512
#define W_      512
#define HW_     (H_*W_)
#define GS_     32
#define NGS_    (GS_*GS_)       // 1024 grid points
#define DW_     11
#define NS_     (DW_*DW_)       // 121 shifts
#define NSP_    128             // shifts padded to WMMA tiles
#define PITCH_  128             // row pitch (floats) for [1024][121] buffers
#define MPITCH_ 136             // LDS mov row pitch in halves (272B, 16B-aligned rows)

typedef __attribute__((ext_vector_type(16))) _Float16 v16h;
typedef __attribute__((ext_vector_type(8)))  float    v8f;

__device__ __forceinline__ void bilin_setup(float cx, float cy,
    int& x0, int& y0, int& x1, int& y1,
    float& w00, float& w01, float& w10, float& w11)
{
    float x = fminf(fmaxf((cx + 1.f) * (0.5f * W_) - 0.5f, 0.f), (float)(W_ - 1));
    float y = fminf(fmaxf((cy + 1.f) * (0.5f * H_) - 0.5f, 0.f), (float)(H_ - 1));
    float xf = floorf(x), yf = floorf(y);
    x0 = (int)xf; y0 = (int)yf;
    x1 = min(x0 + 1, W_ - 1);
    y1 = min(y0 + 1, H_ - 1);
    float wx = x - xf, wy = y - yf;
    w00 = (1.f - wx) * (1.f - wy);
    w01 = wx * (1.f - wy);
    w10 = (1.f - wx) * wy;
    w11 = wx * wy;
}

// ---------------------------------------------------------------------------
// NCHW f32 -> NHWC f16 layout transform. Coalesced reads along W, LDS
// transpose, 16B vector writes of contiguous channel runs.
// grid = (W/32, H), block = 256
// ---------------------------------------------------------------------------
__global__ void __launch_bounds__(256)
nchw2nhwc_kernel(const float* __restrict__ src, _Float16* __restrict__ dst)
{
    __shared__ __align__(16) _Float16 tile[32][128]; // [x][c]
    const int y  = blockIdx.y;
    const int x0 = blockIdx.x * 32;
    const int tid = threadIdx.x;

    const int x  = tid & 31;
    const int c0 = tid >> 5;                 // 0..7
    const float* sp = src + (size_t)y * W_ + x0 + x;
#pragma unroll
    for (int i = 0; i < 16; ++i) {
        int c = c0 + (i << 3);
        tile[x][c] = (_Float16)sp[(size_t)c * HW_];
    }
    __syncthreads();

    const int xx = tid >> 3;                 // 0..31
    const int ch = (tid & 7) << 4;           // chunks of 16 halves (32B)
    uint4 v = *(const uint4*)&tile[xx][ch];
    *(uint4*)(dst + (((size_t)(y * W_ + x0 + xx)) << 7) + ch) = v;
}

// ---------------------------------------------------------------------------
// Per grid point: sample fixed + 121 moving vectors, then
// deeds[s] = a1 + a0 * (||f||^2 + ||m_s||^2 - 2 f.m_s), with the cross term
// computed by v_wmma_f32_16x16x32_f16 (A = 16 shifts x 32 ch, B = fixed
// vector replicated across the 16 columns). One block (4 waves) per point.
// ---------------------------------------------------------------------------
__global__ void __launch_bounds__(128)
deeds_kernel(const float* __restrict__ f00, const float* __restrict__ f50,
             const _Float16* __restrict__ nh00, const _Float16* __restrict__ nh50,
             const float* __restrict__ alpha, float* __restrict__ deeds,
             int use_nhwc)
{
    __shared__ __align__(16) _Float16 movt[NSP_ * MPITCH_];
    __shared__ __align__(16) _Float16 fvec[C_];
    __shared__ float nm[NSP_];
    __shared__ float dotv[NSP_];
    __shared__ float red[128];
    __shared__ float nf_sh;

    const int gs  = blockIdx.x;
    const int tid = threadIdx.x;
    const int gx = gs & 31, gy = gs >> 5;
    const float gxf = (2.f * gx + 1.f) / GS_ - 1.f;
    const float gyf = (2.f * gy + 1.f) / GS_ - 1.f;

    // ---- fixed vector: thread = channel -----------------------------------
    {
        int x0, y0, x1, y1; float w00, w01, w10, w11;
        bilin_setup(gxf, gyf, x0, y0, x1, y1, w00, w01, w10, w11);
        float f;
        if (use_nhwc) {
            const _Float16* p00 = nh00 + (((size_t)(y0 * W_ + x0)) << 7);
            const _Float16* p01 = nh00 + (((size_t)(y0 * W_ + x1)) << 7);
            const _Float16* p10 = nh00 + (((size_t)(y1 * W_ + x0)) << 7);
            const _Float16* p11 = nh00 + (((size_t)(y1 * W_ + x1)) << 7);
            f = w00 * (float)p00[tid] + w01 * (float)p01[tid]
              + w10 * (float)p10[tid] + w11 * (float)p11[tid];
        } else {
            const float* b = f00 + (size_t)tid * HW_;
            f = w00 * b[y0 * W_ + x0] + w01 * b[y0 * W_ + x1]
              + w10 * b[y1 * W_ + x0] + w11 * b[y1 * W_ + x1];
        }
        fvec[tid] = (_Float16)f;
        red[tid]  = f * f;
    }
    __syncthreads();
    for (int s2 = 64; s2 > 0; s2 >>= 1) {
        if (tid < s2) red[tid] += red[tid + s2];
        __syncthreads();
    }
    if (tid == 0) nf_sh = red[0];

    // ---- moving vectors: thread = shift; uint4 (8-half) taps --------------
    {
        const int s = tid;
        float acc = 0.f;
        _Float16* mrowp = movt + s * MPITCH_;
        if (s < NS_) {
            float sx = 0.25f * ((2.f * (s % DW_) + 1.f) / DW_ - 1.f);
            float sy = 0.25f * ((2.f * (s / DW_) + 1.f) / DW_ - 1.f);
            int x0, y0, x1, y1; float w00, w01, w10, w11;
            bilin_setup(gxf + sx, gyf + sy, x0, y0, x1, y1, w00, w01, w10, w11);
            if (use_nhwc) {
                const _Float16* p00 = nh50 + (((size_t)(y0 * W_ + x0)) << 7);
                const _Float16* p01 = nh50 + (((size_t)(y0 * W_ + x1)) << 7);
                const _Float16* p10 = nh50 + (((size_t)(y1 * W_ + x0)) << 7);
                const _Float16* p11 = nh50 + (((size_t)(y1 * W_ + x1)) << 7);
                union Q { uint4 q; _Float16 h[8]; };
                for (int c = 0; c < C_; c += 8) {
                    Q q00, q01, q10, q11, qo;
                    q00.q = *(const uint4*)(p00 + c);
                    q01.q = *(const uint4*)(p01 + c);
                    q10.q = *(const uint4*)(p10 + c);
                    q11.q = *(const uint4*)(p11 + c);
#pragma unroll
                    for (int j = 0; j < 8; ++j) {
                        float m = w00 * (float)q00.h[j] + w01 * (float)q01.h[j]
                                + w10 * (float)q10.h[j] + w11 * (float)q11.h[j];
                        qo.h[j] = (_Float16)m;
                        acc += m * m;
                    }
                    *(uint4*)(mrowp + c) = qo.q;
                }
            } else {
                const int i00 = y0 * W_ + x0, i01 = y0 * W_ + x1;
                const int i10 = y1 * W_ + x0, i11 = y1 * W_ + x1;
                for (int c = 0; c < C_; ++c) {
                    const float* bp = f50 + (size_t)c * HW_;
                    float m = w00 * bp[i00] + w01 * bp[i01]
                            + w10 * bp[i10] + w11 * bp[i11];
                    mrowp[c] = (_Float16)m;
                    acc += m * m;
                }
            }
        } else {
            const uint4 z = {0, 0, 0, 0};
            for (int c = 0; c < C_; c += 8) *(uint4*)(mrowp + c) = z;
        }
        nm[s] = acc;
    }
    __syncthreads();

    // ---- WMMA cross term: dot[s] = fvec . movt[s] -------------------------
    // ISA 7.12.2 f16 layouts are contiguous per lane:
    //   A: two 8-half runs at kb+(hi?8:0) and kb+16+(hi?8:0)
    //   B: two 8-half runs at fvec + kb+(hi?8:0) and kb+16+(hi?8:0)   (column-replicated f)
    {
        const int wave = tid >> 5;
        const int lane = tid & 31;
        const int mrow = lane & 15;
        const bool hi  = lane >= 16;
        const int aoff = hi ? 8 : 0;
        const int boff = hi ? 8 : 0;
        union F { v16h v; uint4 q[2]; };
        for (int t = wave * 2; t < wave * 2 + 2; ++t) {
            v8f acc = {};
            const _Float16* arow0 = movt + (t * 16 + mrow) * MPITCH_;
#pragma unroll
            for (int kb = 0; kb < C_; kb += 32) {
                F A, B;
                A.q[0] = *(const uint4*)(arow0 + kb + aoff);
                A.q[1] = *(const uint4*)(arow0 + kb + 16 + aoff);
                B.q[0] = *(const uint4*)(fvec + kb + boff);
                B.q[1] = *(const uint4*)(fvec + kb + boff + 16);
                acc = __builtin_amdgcn_wmma_f32_16x16x32_f16(
                        false, A.v, false, B.v, (short)0, acc, false, false);
            }
            // D layout: VGPR r, lanes 0-15 -> M=r, lanes 16-31 -> M=8+r;
            // columns are replicated, so lanes 0 and 16 carry all rows.
            if ((lane & 15) == 0) {
                int base = t * 16 + (hi ? 8 : 0);
#pragma unroll
                for (int r = 0; r < 8; ++r) dotv[base + r] = acc[r];
            }
        }
    }
    __syncthreads();

    if (tid < NS_) {
        float d2 = nf_sh + nm[tid] - 2.f * dotv[tid];
        deeds[(size_t)gs * PITCH_ + tid] = alpha[1] + alpha[0] * d2;
    }
}

// ---------------------------------------------------------------------------
// Per grid point: pad3(edge) -> min3 -> avg3 -> avg3 on the 11x11 shift grid.
// second==1: input = a4 + a2*deeds + a3*cavg.
// ---------------------------------------------------------------------------
__global__ void __launch_bounds__(128)
minconv_kernel(const float* __restrict__ deeds, const float* __restrict__ cavg,
               float* __restrict__ outc, const float* __restrict__ alpha, int second)
{
    __shared__ float p[17 * 17];
    __shared__ float q[15 * 15];
    __shared__ float r2[13 * 13];
    const int gs = blockIdx.x, tid = threadIdx.x;
    const float* drow = deeds + (size_t)gs * PITCH_;
    const float* crow = (second && cavg) ? (cavg + (size_t)gs * PITCH_) : drow;
    float a2 = 0.f, a3 = 0.f, a4 = 0.f;
    if (second) { a2 = alpha[2]; a3 = alpha[3]; a4 = alpha[4]; }

    for (int i = tid; i < 17 * 17; i += 128) {
        int yi = i / 17, xi = i % 17;
        int sy = min(max(yi - 3, 0), DW_ - 1);
        int sx = min(max(xi - 3, 0), DW_ - 1);
        int idx = sy * DW_ + sx;
        p[i] = second ? (a4 + a2 * drow[idx] + a3 * crow[idx]) : drow[idx];
    }
    __syncthreads();
    for (int i = tid; i < 15 * 15; i += 128) {
        int y = i / 15, x = i % 15;
        float m = 1e30f;
#pragma unroll
        for (int dy = 0; dy < 3; ++dy)
#pragma unroll
            for (int dx = 0; dx < 3; ++dx)
                m = fminf(m, p[(y + dy) * 17 + (x + dx)]);
        q[i] = m;
    }
    __syncthreads();
    for (int i = tid; i < 13 * 13; i += 128) {
        int y = i / 13, x = i % 13;
        float s = 0.f;
#pragma unroll
        for (int dy = 0; dy < 3; ++dy)
#pragma unroll
            for (int dx = 0; dx < 3; ++dx)
                s += q[(y + dy) * 15 + (x + dx)];
        r2[i] = s * (1.f / 9.f);
    }
    __syncthreads();
    for (int i = tid; i < NS_; i += 128) {
        int y = i / DW_, x = i % DW_;
        float s = 0.f;
#pragma unroll
        for (int dy = 0; dy < 3; ++dy)
#pragma unroll
            for (int dx = 0; dx < 3; ++dx)
                s += r2[(y + dy) * 13 + (x + dx)];
        outc[(size_t)gs * PITCH_ + i] = s * (1.f / 9.f);
    }
}

// ---------------------------------------------------------------------------
// Per shift channel: pad2(edge) -> avg3 -> avg3 over the 32x32 grid of points.
// grid = 121 blocks.
// ---------------------------------------------------------------------------
__global__ void __launch_bounds__(128)
spatial_kernel(const float* __restrict__ cost, float* __restrict__ cavg)
{
    __shared__ float p[36 * 36];
    __shared__ float q[34 * 34];
    const int s = blockIdx.x, tid = threadIdx.x;

    for (int i = tid; i < 36 * 36; i += 128) {
        int yi = i / 36, xi = i % 36;
        int cy = min(max(yi - 2, 0), GS_ - 1);
        int cx = min(max(xi - 2, 0), GS_ - 1);
        p[i] = cost[(size_t)(cy * GS_ + cx) * PITCH_ + s];
    }
    __syncthreads();
    for (int i = tid; i < 34 * 34; i += 128) {
        int y = i / 34, x = i % 34;
        float a = 0.f;
#pragma unroll
        for (int dy = 0; dy < 3; ++dy)
#pragma unroll
            for (int dx = 0; dx < 3; ++dx)
                a += p[(y + dy) * 36 + (x + dx)];
        q[i] = a * (1.f / 9.f);
    }
    __syncthreads();
    for (int i = tid; i < NGS_; i += 128) {
        int y = i / GS_, x = i % GS_;
        float a = 0.f;
#pragma unroll
        for (int dy = 0; dy < 3; ++dy)
#pragma unroll
            for (int dx = 0; dx < 3; ++dx)
                a += q[(y + dy) * 34 + (x + dx)];
        cavg[(size_t)i * PITCH_ + s] = a * (1.f / 9.f);
    }
}

// ---------------------------------------------------------------------------
// Per grid point: softmax(-a5 * cavg2) over 121 shifts + pred_xy = p @ s.
// out = [cost_soft (1024x121) | pred_xy (1024x2)]
// ---------------------------------------------------------------------------
__global__ void __launch_bounds__(128)
softpred_kernel(const float* __restrict__ cavg2, const float* __restrict__ alpha,
                float* __restrict__ out)
{
    __shared__ float red[128];
    const int gs = blockIdx.x, tid = threadIdx.x;
    const float a5 = alpha[5];

    float z = (tid < NS_) ? (-a5 * cavg2[(size_t)gs * PITCH_ + tid]) : -1e30f;
    red[tid] = z; __syncthreads();
    for (int s2 = 64; s2 > 0; s2 >>= 1) {
        if (tid < s2) red[tid] = fmaxf(red[tid], red[tid + s2]);
        __syncthreads();
    }
    const float mx = red[0]; __syncthreads();

    float e = (tid < NS_) ? __expf(z - mx) : 0.f;
    red[tid] = e; __syncthreads();
    for (int s2 = 64; s2 > 0; s2 >>= 1) {
        if (tid < s2) red[tid] += red[tid + s2];
        __syncthreads();
    }
    const float sum = red[0]; __syncthreads();

    const float pr = e / sum;
    if (tid < NS_) out[(size_t)gs * NS_ + tid] = pr;

    float sx = 0.f, sy = 0.f;
    if (tid < NS_) {
        sx = 0.25f * ((2.f * (tid % DW_) + 1.f) / DW_ - 1.f);
        sy = 0.25f * ((2.f * (tid / DW_) + 1.f) / DW_ - 1.f);
    }
    red[tid] = pr * sx; __syncthreads();
    for (int s2 = 64; s2 > 0; s2 >>= 1) {
        if (tid < s2) red[tid] += red[tid + s2];
        __syncthreads();
    }
    const float px = red[0]; __syncthreads();
    red[tid] = pr * sy; __syncthreads();
    for (int s2 = 64; s2 > 0; s2 >>= 1) {
        if (tid < s2) red[tid] += red[tid + s2];
        __syncthreads();
    }
    if (tid == 0) {
        out[(size_t)NGS_ * NS_ + gs * 2 + 0] = px;
        out[(size_t)NGS_ * NS_ + gs * 2 + 1] = red[0];
    }
}

// ---------------------------------------------------------------------------
extern "C" void kernel_launch(void* const* d_in, const int* in_sizes, int n_in,
                              void* d_out, int out_size, void* d_ws, size_t ws_size,
                              hipStream_t stream)
{
    (void)in_sizes; (void)n_in; (void)out_size;
    const float* feat00 = (const float*)d_in[0];
    const float* feat50 = (const float*)d_in[1];
    const float* alpha  = (const float*)d_in[2];
    float* out = (float*)d_out;
    char*  ws  = (char*)d_ws;

    const size_t BUF = (size_t)NGS_ * PITCH_ * sizeof(float);   // 512 KB
    float* deeds = (float*)(ws + 0 * BUF);
    float* cost1 = (float*)(ws + 1 * BUF);
    float* cavgA = (float*)(ws + 2 * BUF);
    float* cost2 = (float*)(ws + 3 * BUF);
    float* cavgB = (float*)(ws + 4 * BUF);
    const size_t small = 5 * BUF;
    const size_t nhB = (size_t)C_ * H_ * W_ * sizeof(_Float16); // 64 MB each
    const int use_nhwc = (ws_size >= small + 2 * nhB) ? 1 : 0;
    _Float16* nh00 = (_Float16*)(ws + small);
    _Float16* nh50 = (_Float16*)(ws + small + nhB);

    if (use_nhwc) {
        dim3 g(W_ / 32, H_);
        nchw2nhwc_kernel<<<g, 256, 0, stream>>>(feat00, nh00);
        nchw2nhwc_kernel<<<g, 256, 0, stream>>>(feat50, nh50);
    }

    deeds_kernel<<<NGS_, 128, 0, stream>>>(feat00, feat50, nh00, nh50,
                                           alpha, deeds, use_nhwc);
    minconv_kernel<<<NGS_, 128, 0, stream>>>(deeds, nullptr, cost1, alpha, 0);
    spatial_kernel<<<NS_, 128, 0, stream>>>(cost1, cavgA);
    minconv_kernel<<<NGS_, 128, 0, stream>>>(deeds, cavgA, cost2, alpha, 1);
    spatial_kernel<<<NS_, 128, 0, stream>>>(cost2, cavgB);
    softpred_kernel<<<NGS_, 128, 0, stream>>>(cavgB, alpha, out);
}
